// GNNFeatureExtractor_6992206757989
// MI455X (gfx1250) — compile-verified, matrix operand order checked by reading, not run on hardware
//
#include <hip/hip_runtime.h>

typedef __attribute__((ext_vector_type(2)))  float    v2f;
typedef __attribute__((ext_vector_type(8)))  float    v8f;
typedef __attribute__((ext_vector_type(16))) _Float16 v16h;

#define DIM 128
#define TM  4            // row tiles (of 16) per block in GEMM
#define LDAP (DIM + 4)   // LDS row padding: stride 132 floats -> no bank conflicts

__device__ __forceinline__ void atomic_add_f32(float* p, float v) {
    __hip_atomic_fetch_add(p, v, __ATOMIC_RELAXED, __HIP_MEMORY_SCOPE_AGENT);
}

// ---------------- elementwise helpers ----------------
__global__ void k_copy4(float4* __restrict__ dst, const float4* __restrict__ src, int n4) {
    int i = blockIdx.x * blockDim.x + threadIdx.x;
    if (i < n4) dst[i] = src[i];
}

__global__ void k_zero(float* __restrict__ p, int n) {
    int i = blockIdx.x * blockDim.x + threadIdx.x;
    if (i < n) p[i] = 0.0f;
}

// ---------------- edge scatter-add: tmp[dst] += h[src]  (one wave per edge) ----------------
__global__ __launch_bounds__(256) void k_scatter(float* __restrict__ tmp,
                                                 const float* __restrict__ h,
                                                 const int* __restrict__ src,
                                                 const int* __restrict__ dst,
                                                 int e) {
    int wave = blockIdx.x * (blockDim.x >> 5) + (threadIdx.x >> 5);
    if (wave >= e) return;                       // wave-uniform branch
    int lane = threadIdx.x & 31;
    int s = src[wave], d = dst[wave];
    float4 v = ((const float4*)(h + (size_t)s * DIM))[lane];   // 512B coalesced row gather
    float* q = tmp + (size_t)d * DIM + lane * 4;
    atomic_add_f32(q + 0, v.x);
    atomic_add_f32(q + 1, v.y);
    atomic_add_f32(q + 2, v.z);
    atomic_add_f32(q + 3, v.w);
}

// ---------------- GEMM: out[N,128] = A[N,128] @ W[128,128] + bias ----------------
// Requires n to be a multiple of 16 (holds here: N = 50000 = 3125 * 16).
__global__ __launch_bounds__(256) void k_gemm_bias(float* __restrict__ out,
                                                   const float* __restrict__ Ain,
                                                   const float* __restrict__ W,
                                                   const float* __restrict__ bias,
                                                   int n, int nRowTiles) {
    __shared__ float As[TM * 16][LDAP];          // 64x(128+4) fp32 strip, ~33.8KB
    const int tid  = threadIdx.x;
    const int wave = tid >> 5;                   // 0..7 -> column tile
    const int lane = tid & 31;
    const int rt0  = blockIdx.x * TM;

    // stage 64x128 A strip into LDS (clamped rows), float4 coalesced
#pragma unroll
    for (int it = 0; it < 8; ++it) {
        int idx  = it * 256 + tid;               // 0..2047 float4 slots
        int r    = idx >> 5;                     // 0..63
        int c4   = idx & 31;                     // 0..31
        int grow = rt0 * 16 + r;
        if (grow > n - 1) grow = n - 1;          // clamp keeps loads in-bounds
        float4 v = ((const float4*)(Ain + (size_t)grow * DIM))[c4];
        ((float4*)&As[r][0])[c4] = v;
    }
    __syncthreads();

    const int colBase = wave * 16;
    const int m  = lane & 15;
    const int hi = lane >> 4;

    v8f acc[TM] = {};

#if __has_builtin(__builtin_amdgcn_wmma_f32_16x16x4_f32)
    // fp32 WMMA path: K stepped by 4, A frag = v2f per lane, B frag = v2f per lane
    const int kk = hi * 2;
    for (int k0 = 0; k0 < DIM; k0 += 4) {
        v2f b;
        b.x = W[(size_t)(k0 + kk)     * DIM + colBase + m];
        b.y = W[(size_t)(k0 + kk + 1) * DIM + colBase + m];
#pragma unroll
        for (int t = 0; t < TM; ++t) {
            v2f a;
            a.x = As[t * 16 + m][k0 + kk];
            a.y = As[t * 16 + m][k0 + kk + 1];
            acc[t] = __builtin_amdgcn_wmma_f32_16x16x4_f32(
                false, a, false, b, (short)0, acc[t], false, false);
        }
    }
#else
    // fallback: f16 inputs, f32 accumulate (codegen-confirmed builtin)
    for (int k0 = 0; k0 < DIM; k0 += 32) {
        v16h b;
#pragma unroll
        for (int j = 0; j < 8; ++j) {
            int kb = k0 + (j & 3) * 2 + hi * 8 + ((j >> 2) << 4);
            b[2 * j]     = (_Float16)W[(size_t)kb       * DIM + colBase + m];
            b[2 * j + 1] = (_Float16)W[(size_t)(kb + 1) * DIM + colBase + m];
        }
#pragma unroll
        for (int t = 0; t < TM; ++t) {
            v16h a;
#pragma unroll
            for (int j = 0; j < 8; ++j) {
                int kb = k0 + (j & 3) * 2 + hi * 8 + ((j >> 2) << 4);
                a[2 * j]     = (_Float16)As[t * 16 + m][kb];
                a[2 * j + 1] = (_Float16)As[t * 16 + m][kb + 1];
            }
            acc[t] = __builtin_amdgcn_wmma_f32_16x16x32_f16(
                false, a, false, b, (short)0, acc[t], false, false);
        }
    }
#endif

    // epilogue: C layout -> lane col = m, VGPR v row = v + 8*hi within tile.
    // Tile-granular guard only; stores are unguarded pointer-strided (n % 16 == 0).
    const float bb = bias[colBase + m];
#pragma unroll
    for (int t = 0; t < TM; ++t) {
        int rt = rt0 + t;
        if (rt < nRowTiles) {
            float* po = out + (size_t)(rt * 16 + 8 * hi) * DIM + colBase + m;
#pragma unroll
            for (int v = 0; v < 8; ++v)
                po[(size_t)v * DIM] = acc[t][v] + bb;
        }
    }
}

// ---------------- BN statistics: per-column sum / sumsq ----------------
__global__ __launch_bounds__(128) void k_bn_stats(const float* __restrict__ z,
                                                  float* __restrict__ stats,
                                                  int rowsPerBlock, int n) {
    int c  = threadIdx.x;                        // 0..127, one column per thread
    int r0 = blockIdx.x * rowsPerBlock;
    int r1 = r0 + rowsPerBlock; if (r1 > n) r1 = n;
    float s = 0.f, q = 0.f;
    for (int r = r0; r < r1; ++r) {
        float v = z[(size_t)r * DIM + c];        // coalesced across threads
        s += v; q += v * v;
    }
    atomic_add_f32(&stats[c], s);
    atomic_add_f32(&stats[DIM + c], q);
}

__global__ __launch_bounds__(128) void k_bn_finalize(float* __restrict__ stats,
                                                     const float* __restrict__ g,
                                                     const float* __restrict__ be,
                                                     float inv_n) {
    int c = threadIdx.x;
    float mean = stats[c] * inv_n;
    float var  = stats[DIM + c] * inv_n - mean * mean;
    float sc   = g[c] * rsqrtf(var + 1e-5f);
    stats[2 * DIM + c] = sc;                     // scale
    stats[3 * DIM + c] = be[c] - mean * sc;      // shift
}

// ---------------- fused scale/shift + ReLU ----------------
__global__ void k_bn_relu(float4* __restrict__ out, const float4* __restrict__ in,
                          const float* __restrict__ stats, int n4) {
    int i = blockIdx.x * blockDim.x + threadIdx.x;
    if (i >= n4) return;
    int c = (i & ((DIM / 4) - 1)) * 4;
    const float* sc = stats + 2 * DIM;
    const float* sh = stats + 3 * DIM;
    float4 v = in[i];
    v.x = fmaxf(0.f, v.x * sc[c + 0] + sh[c + 0]);
    v.y = fmaxf(0.f, v.y * sc[c + 1] + sh[c + 1]);
    v.z = fmaxf(0.f, v.z * sc[c + 2] + sh[c + 2]);
    v.w = fmaxf(0.f, v.w * sc[c + 3] + sh[c + 3]);
    out[i] = v;
}

// ---------------- global_add_pool: out[batch[n]] += h[n]  (one wave per node) ----------------
__global__ __launch_bounds__(256) void k_pool(float* __restrict__ out,
                                              const float* __restrict__ h,
                                              const int* __restrict__ batch,
                                              int n) {
    int wave = blockIdx.x * (blockDim.x >> 5) + (threadIdx.x >> 5);
    if (wave >= n) return;
    int lane = threadIdx.x & 31;
    int g = batch[wave];
    float4 v = ((const float4*)(h + (size_t)wave * DIM))[lane];
    float* q = out + (size_t)g * DIM + lane * 4;
    atomic_add_f32(q + 0, v.x);
    atomic_add_f32(q + 1, v.y);
    atomic_add_f32(q + 2, v.z);
    atomic_add_f32(q + 3, v.w);
}

extern "C" void kernel_launch(void* const* d_in, const int* in_sizes, int n_in,
                              void* d_out, int out_size, void* d_ws, size_t ws_size,
                              hipStream_t stream) {
    const float* x    = (const float*)d_in[0];
    const int*   ei   = (const int*)  d_in[1];
    const int*   batch= (const int*)  d_in[2];
    const float* W1   = (const float*)d_in[3];
    const float* b1   = (const float*)d_in[4];
    const float* g1   = (const float*)d_in[5];
    const float* be1  = (const float*)d_in[6];
    const float* W2   = (const float*)d_in[7];
    const float* b2   = (const float*)d_in[8];
    const float* g2   = (const float*)d_in[9];
    const float* be2  = (const float*)d_in[10];
    float* out = (float*)d_out;

    const int n  = in_sizes[0] / DIM;          // 50000 nodes
    const int e  = in_sizes[1] / 2;            // 800000 edges
    const int Lw = in_sizes[3] / (DIM * DIM);  // 3 layers
    const size_t nd = (size_t)n * DIM;
    const int n4 = (int)(nd / 4);

    float* h     = (float*)d_ws;
    float* tmp   = h + nd;
    float* z     = tmp + nd;
    float* stats = z + nd;                     // 512 floats: sum|sumsq|scale|shift

    const int nRowTiles  = n >> 4;             // full 16-row tiles (n % 16 == 0)
    const int gemmBlocks = (nRowTiles + TM - 1) / TM;
    const int statBlocks = 100;
    const int rowsPerBlk = (n + statBlocks - 1) / statBlocks;
    const float inv_n = 1.0f / (float)n;
    const int eb = (n4 + 255) / 256;

    // h = x
    k_copy4<<<eb, 256, 0, stream>>>((float4*)h, (const float4*)x, n4);

    for (int i = 0; i < Lw; ++i) {
        // tmp = h + sum_{src->dst} h[src]
        k_copy4<<<eb, 256, 0, stream>>>((float4*)tmp, (const float4*)h, n4);
        k_scatter<<<(e + 7) / 8, 256, 0, stream>>>(tmp, h, ei, ei + e, e);
        // z = tmp @ W1 + b1
        k_gemm_bias<<<gemmBlocks, 256, 0, stream>>>(z, tmp, W1 + (size_t)i * DIM * DIM,
                                                    b1 + i * DIM, n, nRowTiles);
        // BN1 + ReLU (in place)
        k_zero<<<1, 256, 0, stream>>>(stats, 2 * DIM);
        k_bn_stats<<<statBlocks, 128, 0, stream>>>(z, stats, rowsPerBlk, n);
        k_bn_finalize<<<1, 128, 0, stream>>>(stats, g1 + i * DIM, be1 + i * DIM, inv_n);
        k_bn_relu<<<eb, 256, 0, stream>>>((float4*)z, (const float4*)z, stats, n4);
        // tmp = z @ W2 + b2
        k_gemm_bias<<<gemmBlocks, 256, 0, stream>>>(tmp, z, W2 + (size_t)i * DIM * DIM,
                                                    b2 + i * DIM, n, nRowTiles);
        // BN2 + ReLU -> h
        k_zero<<<1, 256, 0, stream>>>(stats, 2 * DIM);
        k_bn_stats<<<statBlocks, 128, 0, stream>>>(tmp, stats, rowsPerBlk, n);
        k_bn_finalize<<<1, 128, 0, stream>>>(stats, g2 + i * DIM, be2 + i * DIM, inv_n);
        k_bn_relu<<<eb, 256, 0, stream>>>((float4*)h, (const float4*)tmp, stats, n4);
    }

    // out = segment_sum(h, batch)
    k_zero<<<(out_size + 255) / 256, 256, 0, stream>>>(out, out_size);
    k_pool<<<(n + 7) / 8, 256, 0, stream>>>(out, h, batch, n);
}
